// TemporalGAT_16028817948757
// MI455X (gfx1250) — matrix operand order repaired and checked by reading, not compiled, not run on hardware
//
#include <hip/hip_runtime.h>

typedef __attribute__((ext_vector_type(16))) __bf16 v16bf;
typedef __attribute__((ext_vector_type(8)))  float  v8f;

__device__ inline unsigned short f2bf(float f){
  unsigned u = __float_as_uint(f);
  unsigned r = u + 0x7FFFu + ((u >> 16) & 1u);
  return (unsigned short)(r >> 16);
}

__device__ inline unsigned mapf(float f){
  unsigned b = __float_as_uint(f);
  return (b & 0x80000000u) ? ~b : (b | 0x80000000u);
}
__device__ inline float unmapf(unsigned u){
  unsigned b = (u & 0x80000000u) ? (u & 0x7FFFFFFFu) : ~u;
  return __uint_as_float(b);
}

__device__ inline v8f wmma_bf(v16bf a, v16bf b, v8f c){
  return __builtin_amdgcn_wmma_f32_16x16x32_bf16(false, a, false, b, (short)0, c, false, false);
}

// Defeat LICM while PRESERVING pointer provenance (so loads stay global_load, not flat_load):
// launder a zero offset through an opaque asm, then GEP off the original kernarg pointer.
__device__ inline int opaque_zero(){
  int v = 0;
  asm volatile("" : "+s"(v));
  return v;
}

// Swizzled tile: 512 bf16 = one 16x32 (A) or 32x16 (B) fragment, stored [lane][16 elems].
// A element (row m, k kk): lane = ((kk>>3)&1)*16 + m ; elem = ((kk>>4)&1)*8 + (kk&7)
// B element (k kk, col):   lane = ((kk>>3)&1)*16 + col ; elem = ((kk>>4)&1)*8 + (kk&7)
__device__ inline int a_swz_idx(int m, int kk){
  int lane = (((kk >> 3) & 1) << 4) + m;
  int elem = (((kk >> 4) & 1) << 3) + (kk & 7);
  return lane * 16 + elem;
}

__device__ inline v16bf load_tile(const unsigned short* T){
  return *(const v16bf*)(T + (threadIdx.x & 31) * 16);
}

// C/D f32 16x16: VGPR r, lane L -> row r + 8*(L>>4), col L&15
__device__ inline void store_C_lds(float* S, int ld, int n0, v8f c){
  int lane = threadIdx.x & 31;
  int half = lane >> 4, col = lane & 15;
#pragma unroll
  for (int r = 0; r < 8; r++) S[(r + 8 * half) * ld + n0 + col] = c[r];
}

// Swizzle one bf16 element of a B-layout weight. i indexes the destination (Kpad*N elems).
// trans: source is [N][K] (for W.T GEMMs); else [K][N].
__device__ inline void swz_w(const float* W, unsigned short* D, int K, int N, bool trans, int i){
  int ntiles = N >> 4;
  int tile = i >> 9, r = i & 511;
  int lane = r >> 4, elem = r & 15;
  int col = lane & 15, half = lane >> 4;
  int kk = ((elem >> 3) << 4) + half * 8 + (elem & 7);
  int nt = tile % ntiles, kc = tile / ntiles;
  int k = kc * 32 + kk;
  float v = 0.f;
  if (k < K) v = trans ? W[(long)(nt * 16 + col) * K + k] : W[(long)k * N + nt * 16 + col];
  D[i] = f2bf(v);
}

__global__ void prep_weights(const float* gwih, const float* gwhh, const float* ainw,
                             const float* aoutw, const float* projw, const float* g1w, const float* g2w,
                             unsigned short* wihT, unsigned short* whhT, unsigned short* inT,
                             unsigned short* outT, unsigned short* projT,
                             unsigned short* g1, unsigned short* g2){
  int i = blockIdx.x * blockDim.x + threadIdx.x;
  if (i < 32 * 192) swz_w(gwih, wihT, 16, 192, true, i);     // K=16 zero-padded to 32
  if (i < 64 * 192){ swz_w(gwhh, whhT, 64, 192, true, i);
                     swz_w(ainw, inT,  64, 192, true, i); }
  if (i < 64 * 64){ swz_w(aoutw, outT, 64, 64, true, i);
                    swz_w(projw, projT, 64, 64, true, i); }
  if (i < 128 * 256) swz_w(g1w, g1, 128, 256, false, i);
  if (i < 256 * 256) swz_w(g2w, g2, 256, 256, false, i);
}

// ---------------- fused GRU: 1 wave / 16 nodes; writes gru_out in swizzled tile layout ----------------
__global__ void gru_kernel(const float* __restrict__ seq,
                           const unsigned short* __restrict__ wihT, const unsigned short* __restrict__ whhT,
                           const float* __restrict__ bih, const float* __restrict__ bhh,
                           unsigned short* __restrict__ gru_swz, float* __restrict__ last_h, int n){
  __shared__ __align__(32) unsigned short seq_swz[24 * 512];
  __shared__ __align__(32) unsigned short h_swz[2 * 512];
  __shared__ float h_f[16 * 64];
  __shared__ float sc[16 * 256]; // 0..127: gi+gh (r,z); 128..191: gi_n; 192..255: gh_n
  int lane = threadIdx.x;
  long node0 = (long)blockIdx.x * 16;

  // fill seq tiles pre-swizzled (K padded 16->32 with zeros)
  for (int i = lane; i < 24 * 512; i += 32){
    int t = i >> 9, r = i & 511;
    int ld = r >> 4, elem = r & 15;
    int m = ld & 15, hb = ld >> 4;
    int kk = ((elem >> 3) << 4) | (hb << 3) | (elem & 7);
    long node = node0 + m; if (node >= n) node = n - 1;
    float v = (kk < 16) ? seq[node * 384 + t * 16 + kk] : 0.f;
    seq_swz[i] = f2bf(v);
  }
  for (int i = lane; i < 1024; i += 32){ h_f[i] = 0.f; h_swz[i] = 0; }
  __syncthreads();

  for (int t = 0; t < 24; t++){
    // opaque offset: per-iteration fragment loads (global, L0 hits) instead of hoist + spill
    int zo = opaque_zero();
    const unsigned short* wp = wihT + zo;
    const unsigned short* hp = whhT + zo;
    v16bf a_seq = load_tile(seq_swz + t * 512);
    v16bf a_h0  = load_tile(h_swz);
    v16bf a_h1  = load_tile(h_swz + 512);
#pragma unroll
    for (int nt = 0; nt < 8; nt++){ // r,z combined: gi + gh
      v8f c = {};
      c = wmma_bf(a_seq, load_tile(wp + nt * 512), c);
      c = wmma_bf(a_h0,  load_tile(hp + nt * 512), c);
      c = wmma_bf(a_h1,  load_tile(hp + (12 + nt) * 512), c);
      store_C_lds(sc, 256, nt * 16, c);
    }
#pragma unroll
    for (int nt = 0; nt < 4; nt++){ // gi_n
      v8f c = {};
      c = wmma_bf(a_seq, load_tile(wp + (8 + nt) * 512), c);
      store_C_lds(sc, 256, 128 + nt * 16, c);
    }
#pragma unroll
    for (int nt = 0; nt < 4; nt++){ // gh_n
      v8f c = {};
      c = wmma_bf(a_h0, load_tile(hp + (8 + nt) * 512), c);
      c = wmma_bf(a_h1, load_tile(hp + (20 + nt) * 512), c);
      store_C_lds(sc, 256, 192 + nt * 16, c);
    }
    __syncthreads();
    unsigned short* gdst = gru_swz + ((long)blockIdx.x * 24 + t) * 1024;
    for (int i = lane; i < 1024; i += 32){
      int m = i >> 6, j = i & 63;
      float r = sc[m * 256 + j]      + bih[j]      + bhh[j];
      float z = sc[m * 256 + 64 + j] + bih[64 + j] + bhh[64 + j];
      r = 1.f / (1.f + __expf(-r));
      z = 1.f / (1.f + __expf(-z));
      float gin = sc[m * 256 + 128 + j] + bih[128 + j];
      float ghn = sc[m * 256 + 192 + j] + bhh[128 + j];
      float nn = tanhf(gin + r * ghn);
      float hn = (1.f - z) * nn + z * h_f[i];
      h_f[i] = hn;
      unsigned short hb2 = f2bf(hn);
      int pos = ((j >> 5) << 9) + a_swz_idx(m, j & 31);
      h_swz[pos] = hb2;
      gdst[pos] = hb2;
    }
    __syncthreads();
  }
  for (int i = lane; i < 1024; i += 32){
    int m = i >> 6, j = i & 63;
    long node = node0 + m;
    if (node < n) last_h[node * 64 + j] = h_f[i];
  }
}

// ---------------- MHA (last query) + proj + xin (swizzled) : 1 wave / 16 nodes ----------------
// inT tiles: q: kc*12 + 0..3 ; k: kc*12 + 4..7 ; v: kc*12 + 8..11  (kc = 0,1)
__global__ void mha_kernel(const unsigned short* __restrict__ gsw, const float* __restrict__ last_h,
                           const float* __restrict__ x,
                           const unsigned short* __restrict__ inT, const float* __restrict__ in_b,
                           const unsigned short* __restrict__ outT, const float* __restrict__ out_b,
                           const unsigned short* __restrict__ projT, const float* __restrict__ proj_b,
                           unsigned short* __restrict__ xin_swz, int n){
  __shared__ float q_lds[16 * 64];
  __shared__ float kt[16 * 64];
  __shared__ float scx[16 * 4 * 24];
  __shared__ float o[16 * 64];
  __shared__ __align__(32) unsigned short abf[2 * 512];
  int lane = threadIdx.x;
  long node0 = (long)blockIdx.x * 16;
  const unsigned short* gblk = gsw + (long)blockIdx.x * 24 * 1024;

  { // q at t=23
    v16bf a0 = load_tile(gblk + 23 * 1024);
    v16bf a1 = load_tile(gblk + 23 * 1024 + 512);
#pragma unroll
    for (int nt = 0; nt < 4; nt++){
      v8f c = {};
      c = wmma_bf(a0, load_tile(inT + nt * 512), c);
      c = wmma_bf(a1, load_tile(inT + (12 + nt) * 512), c);
      store_C_lds(q_lds, 64, nt * 16, c);
    }
  }
  __syncthreads();
  for (int i = lane; i < 1024; i += 32) q_lds[i] += in_b[i & 63];
  __syncthreads();

  for (int t = 0; t < 24; t++){ // K_t and scores
    const unsigned short* ip = inT + opaque_zero();
    v16bf a0 = load_tile(gblk + t * 1024);
    v16bf a1 = load_tile(gblk + t * 1024 + 512);
#pragma unroll
    for (int nt = 0; nt < 4; nt++){
      v8f c = {};
      c = wmma_bf(a0, load_tile(ip + (4 + nt) * 512), c);
      c = wmma_bf(a1, load_tile(ip + (16 + nt) * 512), c);
      store_C_lds(kt, 64, nt * 16, c);
    }
    __syncthreads();
    for (int p = lane; p < 64; p += 32){
      int m = p >> 2, h = p & 3;
      float s = 0.f;
#pragma unroll
      for (int d = 0; d < 16; d++){
        int j = h * 16 + d;
        s += q_lds[m * 64 + j] * (kt[m * 64 + j] + in_b[64 + j]);
      }
      scx[p * 24 + t] = s * 0.25f; // hd=16 -> 1/sqrt(16)
    }
    __syncthreads();
  }
  for (int p = lane; p < 64; p += 32){ // softmax over t
    float mx = -1e30f;
    for (int t = 0; t < 24; t++) mx = fmaxf(mx, scx[p * 24 + t]);
    float ss = 0.f;
    for (int t = 0; t < 24; t++){ float e = __expf(scx[p * 24 + t] - mx); scx[p * 24 + t] = e; ss += e; }
    float inv = 1.f / ss;
    for (int t = 0; t < 24; t++) scx[p * 24 + t] *= inv;
  }
  for (int i = lane; i < 1024; i += 32) o[i] = 0.f;
  __syncthreads();

  for (int t = 0; t < 24; t++){ // V_t accumulation
    const unsigned short* ip = inT + opaque_zero();
    v16bf a0 = load_tile(gblk + t * 1024);
    v16bf a1 = load_tile(gblk + t * 1024 + 512);
#pragma unroll
    for (int nt = 0; nt < 4; nt++){
      v8f c = {};
      c = wmma_bf(a0, load_tile(ip + (8 + nt) * 512), c);
      c = wmma_bf(a1, load_tile(ip + (20 + nt) * 512), c);
      store_C_lds(kt, 64, nt * 16, c);
    }
    __syncthreads();
    for (int i = lane; i < 1024; i += 32){
      int m = i >> 6, j = i & 63, h = j >> 4;
      o[i] += scx[(m * 4 + h) * 24 + t] * (kt[i] + in_b[128 + j]);
    }
    __syncthreads();
  }
  for (int i = lane; i < 1024; i += 32){
    int m = i >> 6, j = i & 63;
    abf[((j >> 5) << 9) + a_swz_idx(m, j & 31)] = f2bf(o[i]);
  }
  __syncthreads();
  { // out projection (outT: 8 tiles, kc*4+nt)
    v16bf a0 = load_tile(abf);
    v16bf a1 = load_tile(abf + 512);
#pragma unroll
    for (int nt = 0; nt < 4; nt++){
      v8f c = {};
      c = wmma_bf(a0, load_tile(outT + nt * 512), c);
      c = wmma_bf(a1, load_tile(outT + (4 + nt) * 512), c);
      store_C_lds(kt, 64, nt * 16, c);
    }
  }
  __syncthreads();
  for (int i = lane; i < 1024; i += 32){ // last = last_h + attn + out_b  -> abf (swizzled)
    int m = i >> 6, j = i & 63;
    long node = node0 + m; if (node >= n) node = n - 1;
    float lh = last_h[node * 64 + j];
    abf[((j >> 5) << 9) + a_swz_idx(m, j & 31)] = f2bf(lh + kt[i] + out_b[j]);
  }
  __syncthreads();
  { // t_emb = last @ projT
    v16bf a0 = load_tile(abf);
    v16bf a1 = load_tile(abf + 512);
#pragma unroll
    for (int nt = 0; nt < 4; nt++){
      v8f c = {};
      c = wmma_bf(a0, load_tile(projT + nt * 512), c);
      c = wmma_bf(a1, load_tile(projT + (4 + nt) * 512), c);
      store_C_lds(o, 64, nt * 16, c);
    }
  }
  __syncthreads();
  unsigned short* xb = xin_swz + (long)blockIdx.x * 4 * 512;
  for (int i = lane; i < 1024; i += 32){
    int m = i >> 6, j = i & 63;
    long node = node0 + m; if (node >= n) node = n - 1;
    float xv = x[node * 64 + j];
    float tv = o[i] + proj_b[j];
    int pos = a_swz_idx(m, j & 31);
    int c0 = j >> 5;
    xb[c0 * 512 + pos]       = f2bf(xv);
    xb[(2 + c0) * 512 + pos] = f2bf(tv);
  }
}

// ---------------- GAT feature GEMM: swizzled A [blk][KC][512] @ swizzled B -> C f32 [rows][Nout] ----------
template<int KC>
__global__ void gat_gemm(const unsigned short* __restrict__ A, const unsigned short* __restrict__ B,
                         float* __restrict__ C, int rows, int Nout){
  int lane = threadIdx.x & 31;
  int half = lane >> 4, col = lane & 15;
  long m0 = (long)blockIdx.x * 16;
  const unsigned short* ab = A + (long)blockIdx.x * KC * 512;
  v16bf a[KC];
#pragma unroll
  for (int kc = 0; kc < KC; kc++) a[kc] = load_tile(ab + kc * 512);
  int ntiles = Nout >> 4;
  for (int nt = 0; nt < ntiles; nt++){
    v8f c = {};
#pragma unroll
    for (int kc = 0; kc < KC; kc++)
      c = wmma_bf(a[kc], load_tile(B + (long)(kc * ntiles + nt) * 512), c);
#pragma unroll
    for (int r = 0; r < 8; r++){
      long m = m0 + r + 8 * half;
      if (m < rows) C[m * Nout + nt * 16 + col] = c[r];
    }
  }
}

// ---------------- per-node attention logits ----------------
__global__ void att_scores(const float* __restrict__ pre, const float* __restrict__ att_src,
                           const float* __restrict__ att_dst,
                           float* a_src, float* a_dst, int n){
  int i = blockIdx.x * blockDim.x + threadIdx.x;
  if (i >= n * 4) return;
  int node = i >> 2, h = i & 3;
  const float* p = pre + (long)node * 256 + h * 64;
  float s = 0.f, d = 0.f;
#pragma unroll 8
  for (int c = 0; c < 64; c++){ s += p[c] * att_src[h * 64 + c]; d += p[c] * att_dst[h * 64 + c]; }
  a_src[i] = s; a_dst[i] = d;
}

__global__ void init_layer(unsigned* mbits, float* ssum, float* agg, int n){
  int i = blockIdx.x * blockDim.x + threadIdx.x;
  if (i < n * 4){ mbits[i] = mapf(-3.0e38f); ssum[i] = 0.f; }
  if (i < n * 256) agg[i] = 0.f;
}

__global__ void edge_max(const int* __restrict__ ei, int E0, int E, const float* __restrict__ a_src,
                         const float* __restrict__ a_dst, unsigned* mbits){
  int i = blockIdx.x * blockDim.x + threadIdx.x;
  if (i >= E * 4) return;
  int e = i >> 2, h = i & 3;
  int s = (e < E0) ? ei[e] : (e - E0);
  int d = (e < E0) ? ei[E0 + e] : (e - E0);
  float v = a_src[s * 4 + h] + a_dst[d * 4 + h];
  v = (v > 0.f) ? v : 0.2f * v;
  atomicMax(&mbits[d * 4 + h], mapf(v));
}

__global__ void edge_exp(const int* __restrict__ ei, int E0, int E, const float* __restrict__ a_src,
                         const float* __restrict__ a_dst, const unsigned* __restrict__ mbits,
                         float* exv, float* ssum){
  int i = blockIdx.x * blockDim.x + threadIdx.x;
  if (i >= E * 4) return;
  int e = i >> 2, h = i & 3;
  int s = (e < E0) ? ei[e] : (e - E0);
  int d = (e < E0) ? ei[E0 + e] : (e - E0);
  float v = a_src[s * 4 + h] + a_dst[d * 4 + h];
  v = (v > 0.f) ? v : 0.2f * v;
  float ex = __expf(v - unmapf(mbits[d * 4 + h]));
  exv[i] = ex;
  atomicAdd(&ssum[d * 4 + h], ex);
}

__global__ void edge_msg(const int* __restrict__ ei, int E0, long long total, const float* __restrict__ pre,
                         const float* __restrict__ exv, const float* __restrict__ ssum, float* agg){
  long long i = (long long)blockIdx.x * blockDim.x + threadIdx.x;
  if (i >= total) return;
  int c = (int)(i & 255);
  long long e = i >> 8;
  int h = c >> 6;
  int s = (e < E0) ? ei[e] : (int)(e - E0);
  int d = (e < E0) ? ei[E0 + e] : (int)(e - E0);
  float alpha = exv[e * 4 + h] / ssum[d * 4 + h];
  atomicAdd(&agg[(long)d * 256 + c], pre[(long)s * 256 + c] * alpha);
}

// GAT1 finalize: bias + elu -> h1 in swizzled A layout [blk][8][512]
__global__ void gat1_fin(const float* __restrict__ agg, const float* __restrict__ bias,
                         unsigned short* __restrict__ h1swz, int n){
  int i = blockIdx.x * blockDim.x + threadIdx.x;
  if (i >= n * 256) return;
  int c = i & 255, node = i >> 8;
  float v = agg[i] + bias[c];
  v = (v > 0.f) ? v : (__expf(v) - 1.f); // elu
  int blk = node >> 4, m = node & 15;
  int chunk = c >> 5, kk = c & 31;
  h1swz[((long)blk * 8 + chunk) * 512 + a_swz_idx(m, kk)] = f2bf(v);
}

__global__ void final_kernel(const float* __restrict__ agg, const float* __restrict__ b2,
                             const float* __restrict__ linw, const float* __restrict__ linb,
                             float* out, int n){
  __shared__ float red[64];
  int node = blockIdx.x, c = threadIdx.x;
  long b = (long)node * 256;
  float v = 0.25f * (agg[b + c] + agg[b + 64 + c] + agg[b + 128 + c] + agg[b + 192 + c]) + b2[c];
  v = (v > 0.f) ? v : (__expf(v) - 1.f);
  red[c] = v * linw[c];
  __syncthreads();
  for (int s = 32; s > 0; s >>= 1){
    if (c < s) red[c] += red[c + s];
    __syncthreads();
  }
  if (c == 0) out[node] = red[0] + linb[0];
}

extern "C" void kernel_launch(void* const* d_in, const int* in_sizes, int n_in,
                              void* d_out, int out_size, void* d_ws, size_t ws_size,
                              hipStream_t stream) {
  const float* x     = (const float*)d_in[0];
  const int*   ei    = (const int*)d_in[1];
  const float* seq   = (const float*)d_in[2];
  const float* gwih  = (const float*)d_in[3];
  const float* gwhh  = (const float*)d_in[4];
  const float* gbih  = (const float*)d_in[5];
  const float* gbhh  = (const float*)d_in[6];
  const float* ainw  = (const float*)d_in[7];
  const float* ainb  = (const float*)d_in[8];
  const float* aoutw = (const float*)d_in[9];
  const float* aoutb = (const float*)d_in[10];
  const float* projw = (const float*)d_in[11];
  const float* projb = (const float*)d_in[12];
  const float* g1w   = (const float*)d_in[13];
  const float* g1as  = (const float*)d_in[14];
  const float* g1ad  = (const float*)d_in[15];
  const float* g1b   = (const float*)d_in[16];
  const float* g2w   = (const float*)d_in[17];
  const float* g2as  = (const float*)d_in[18];
  const float* g2ad  = (const float*)d_in[19];
  const float* g2b   = (const float*)d_in[20];
  const float* linw  = (const float*)d_in[21];
  const float* linb  = (const float*)d_in[22];

  int n  = in_sizes[0] / 64;
  int E0 = in_sizes[1] / 2;
  int E  = E0 + n;
  int nb16 = (n + 15) / 16;

  char* p = (char*)d_ws;
  auto alloc = [&](size_t bytes) -> char* {
    char* r = p; p += (bytes + 255) & ~(size_t)255; return r;
  };
  unsigned short* wihT   = (unsigned short*)alloc(32 * 192 * 2);
  unsigned short* whhT   = (unsigned short*)alloc(64 * 192 * 2);
  unsigned short* inT    = (unsigned short*)alloc(64 * 192 * 2);
  unsigned short* outT   = (unsigned short*)alloc(64 * 64 * 2);
  unsigned short* projT  = (unsigned short*)alloc(64 * 64 * 2);
  unsigned short* g1     = (unsigned short*)alloc(128 * 256 * 2);
  unsigned short* g2     = (unsigned short*)alloc(256 * 256 * 2);
  unsigned short* gswz   = (unsigned short*)alloc((size_t)nb16 * 24 * 1024 * 2);
  float* last_h          = (float*)alloc((size_t)n * 64 * 4);
  unsigned short* xinswz = (unsigned short*)alloc((size_t)nb16 * 4 * 512 * 2);
  float* pre             = (float*)alloc((size_t)n * 256 * 4);
  float* a_src           = (float*)alloc((size_t)n * 4 * 4);
  float* a_dst           = (float*)alloc((size_t)n * 4 * 4);
  unsigned* mbits        = (unsigned*)alloc((size_t)n * 4 * 4);
  float* ssum            = (float*)alloc((size_t)n * 4 * 4);
  float* exv             = (float*)alloc((size_t)E * 4 * 4);
  float* agg             = (float*)alloc((size_t)n * 256 * 4);
  unsigned short* h1swz  = (unsigned short*)alloc((size_t)nb16 * 8 * 512 * 2);

  int nb_n4   = (n * 4 + 255) / 256;
  int nb_n256 = (n * 256 + 255) / 256;
  int nb_e4   = (E * 4 + 255) / 256;
  long long msg_total = (long long)E * 256;
  int nb_msg  = (int)((msg_total + 255) / 256);

  prep_weights<<<256, 256, 0, stream>>>(gwih, gwhh, ainw, aoutw, projw, g1w, g2w,
                                        wihT, whhT, inT, outT, projT, g1, g2);
  gru_kernel<<<nb16, 32, 0, stream>>>(seq, wihT, whhT, gbih, gbhh, gswz, last_h, n);
  mha_kernel<<<nb16, 32, 0, stream>>>(gswz, last_h, x, inT, ainb, outT, aoutb,
                                      projT, projb, xinswz, n);

  // ---- GAT layer 1 ----
  gat_gemm<4><<<nb16, 32, 0, stream>>>(xinswz, g1, pre, n, 256);
  att_scores<<<nb_n4, 256, 0, stream>>>(pre, g1as, g1ad, a_src, a_dst, n);
  init_layer<<<nb_n256, 256, 0, stream>>>(mbits, ssum, agg, n);
  edge_max<<<nb_e4, 256, 0, stream>>>(ei, E0, E, a_src, a_dst, mbits);
  edge_exp<<<nb_e4, 256, 0, stream>>>(ei, E0, E, a_src, a_dst, mbits, exv, ssum);
  edge_msg<<<nb_msg, 256, 0, stream>>>(ei, E0, msg_total, pre, exv, ssum, agg);
  gat1_fin<<<nb_n256, 256, 0, stream>>>(agg, g1b, h1swz, n);

  // ---- GAT layer 2 ----
  gat_gemm<8><<<nb16, 32, 0, stream>>>(h1swz, g2, pre, n, 256);
  att_scores<<<nb_n4, 256, 0, stream>>>(pre, g2as, g2ad, a_src, a_dst, n);
  init_layer<<<nb_n256, 256, 0, stream>>>(mbits, ssum, agg, n);
  edge_max<<<nb_e4, 256, 0, stream>>>(ei, E0, E, a_src, a_dst, mbits);
  edge_exp<<<nb_e4, 256, 0, stream>>>(ei, E0, E, a_src, a_dst, mbits, exv, ssum);
  edge_msg<<<nb_msg, 256, 0, stream>>>(ei, E0, msg_total, pre, exv, ssum, agg);

  final_kernel<<<n, 64, 0, stream>>>(agg, g2b, linw, linb, (float*)d_out, n);
}
